// NaSwinAttention_10110353015209
// MI455X (gfx1250) — matrix operand order repaired and checked by reading, not compiled, not run on hardware
//
#include <hip/hip_runtime.h>
#include <hip/hip_bf16.h>
#include <stdint.h>
#include <stddef.h>

// ---------------------------------------------------------------------------
// Problem constants (from reference)
// ---------------------------------------------------------------------------
#define T_    8
#define H_    32
#define W_    32
#define WT_   4
#define WH_   8
#define WW_   8
#define HEADS 16
#define HD    128
#define DIM   2048          // HEADS*HD
#define DIM3  6144          // 3*DIM
#define LT    256
#define NWIN  32
#define WLEN  256           // WT*WH*WW
#define SEQ   512           // WLEN + LT
#define LV    8192          // T*H*W

typedef __bf16 bf16_t;
typedef __attribute__((ext_vector_type(16))) __bf16 v16bf;
typedef __attribute__((ext_vector_type(8)))  __bf16 v8bf;
typedef __attribute__((ext_vector_type(8)))  float  v8f;
typedef unsigned int u32x4 __attribute__((ext_vector_type(4)));
typedef int          i32x4 __attribute__((ext_vector_type(4)));
typedef int          i32x8 __attribute__((ext_vector_type(8)));

// ---------------------------------------------------------------------------
// bf16 helpers
// ---------------------------------------------------------------------------
__device__ inline bf16_t f2bf(float f) {
  union { float f; uint32_t u; } x; x.f = f;
  uint32_t r = x.u + 0x7FFFu + ((x.u >> 16) & 1u);
  uint16_t h = (uint16_t)(r >> 16);
  bf16_t o; __builtin_memcpy(&o, &h, 2); return o;
}
__device__ inline float bf2f(bf16_t b) {
  uint16_t h; __builtin_memcpy(&h, &b, 2);
  union { uint32_t u; float f; } x; x.u = ((uint32_t)h) << 16; return x.f;
}
__device__ inline bf16_t to_bf_elem(float v)  { return f2bf(v); }
__device__ inline bf16_t to_bf_elem(bf16_t v) { return v; }
__device__ inline void store_out(bf16_t* p, float v) { *p = f2bf(v); }
__device__ inline void store_out(float*  p, float v) { *p = v; }
__device__ inline v16bf concat8(v8bf lo, v8bf hi) {
  return __builtin_shufflevector(lo, hi, 0,1,2,3,4,5,6,7,8,9,10,11,12,13,14,15);
}

// ---------------------------------------------------------------------------
// Tensor Data Mover: async 2D tile load (bf16 elements) global -> LDS.
// Descriptor bitfields per CDNA5 ISA (D# group0/group1).
//   group0: [1:0]=count=1, [63:32]=lds byte addr, [120:64]=global addr,
//           [127:126]=type=2
//   group1: data_size=1 (2B), tensor_dim0=row_elems, tensor_dim1=rows,
//           tile_dim0=row_elems, tile_dim1=rows, tensor_dim0_stride=stride
// Issued per-wave (EXEC ignored); completion tracked with TENSORcnt.
// ---------------------------------------------------------------------------
__device__ inline void tdm_load_2d_to_lds(uint32_t lds_byte_off, const void* gptr,
                                          uint32_t row_elems, uint32_t rows,
                                          uint32_t row_stride_elems) {
  const uint64_t ga = (uint64_t)(uintptr_t)gptr;
  u32x4 g0;
  g0[0] = 1u;                                                  // count=1
  g0[1] = lds_byte_off;                                        // lds_addr
  g0[2] = (uint32_t)ga;                                        // global_addr lo
  g0[3] = (uint32_t)((ga >> 32) & 0x01FFFFFFu) | (2u << 30);   // addr hi | type=2
  i32x8 g1;
  g1[0] = (int)(1u << 16);                 // workgroup_mask=0, data_size=1 (2B)
  g1[1] = (int)(row_elems << 16);          // tensor_dim0[15:0] in [31:16]
  g1[2] = (int)(rows << 16);               // tensor_dim0 hi=0 | tensor_dim1[15:0]
  g1[3] = (int)(row_elems << 16);          // tensor_dim1 hi=0 | tile_dim0
  g1[4] = (int)rows;                       // tile_dim1 | tile_dim2=0
  g1[5] = (int)row_stride_elems;           // tensor_dim0_stride[31:0]
  g1[6] = 0;                               // stride hi | tensor_dim1_stride lo
  g1[7] = 0;
  i32x4 z4 = {0, 0, 0, 0};
#if defined(__clang_major__) && (__clang_major__ >= 23)
  i32x8 z8 = {0, 0, 0, 0, 0, 0, 0, 0};
  __builtin_amdgcn_tensor_load_to_lds(g0, g1, z4, z4, z8, 0);
#else
  __builtin_amdgcn_tensor_load_to_lds(g0, g1, z4, z4, 0);
#endif
}

// ---------------------------------------------------------------------------
// Generic GEMM:  C[M,N] = A[M,K] * B[K,N] + bias[N]
// A is fp32 or bf16, B is fp32 (converted to bf16 while staging), C is bf16 or
// fp32.  Block tile 128x128, BK=32, 8 waves in 2x4 layout, each wave 64x32 via
// 4x2 tiles of v_wmma_f32_16x16x32_bf16.  M,N multiples of 128; K mult of 32.
// ---------------------------------------------------------------------------
template <typename AT, typename OT>
__global__ __launch_bounds__(256)
void gemm_bias_kernel(const AT* __restrict__ A, const float* __restrict__ B,
                      const float* __restrict__ bias, OT* __restrict__ C,
                      int M, int N, int K) {
  __shared__ __align__(64) bf16_t As[128 * 32];   // [m][k], k contiguous
  __shared__ __align__(64) bf16_t Bst[128 * 32];  // [n][k], k contiguous
  (void)M;

  const int bn0  = blockIdx.x * 128;
  const int bm0  = blockIdx.y * 128;
  const int tid  = threadIdx.x;
  const int wave = tid >> 5;
  const int lane = tid & 31;
  const int wm   = wave >> 2;     // 0..1
  const int wn   = wave & 3;      // 0..3
  const int l16  = lane & 15;
  const int half = lane >> 4;

  v8f acc[4][2];
#pragma unroll
  for (int i = 0; i < 4; ++i)
#pragma unroll
    for (int j = 0; j < 2; ++j)
#pragma unroll
      for (int r = 0; r < 8; ++r) acc[i][j][r] = 0.0f;

  for (int k0 = 0; k0 < K; k0 += 32) {
    // ---- stage A tile (128x32) ----
    {
      const int row = tid >> 1;
      const int ks  = (tid & 1) * 16;
      const AT* src = A + (size_t)(bm0 + row) * K + k0 + ks;
      bf16_t* dst = As + row * 32 + ks;
#pragma unroll
      for (int i = 0; i < 16; ++i) dst[i] = to_bf_elem(src[i]);
    }
    // ---- stage B tile transposed to [n][k] ----
    {
      const int kk = tid >> 3;             // 0..31
      const int n0 = (tid & 7) * 16;       // 0..112
      const float* src = B + (size_t)(k0 + kk) * N + bn0 + n0;
#pragma unroll
      for (int i = 0; i < 16; ++i) Bst[(n0 + i) * 32 + kk] = f2bf(src[i]);
    }
    // prefetch next K-step tiles (global_prefetch_b8; 192MB L2 on MI455X)
    if (k0 + 32 < K) {
      __builtin_prefetch(A + (size_t)(bm0 + (tid >> 1)) * K + k0 + 32, 0, 1);
      __builtin_prefetch(B + (size_t)(k0 + 32 + (tid >> 3)) * N + bn0 + (tid & 7) * 16, 0, 1);
    }
    __syncthreads();

    // ---- fragments ----
    v16bf af[4];
    {
      const int klo = half * 8;
#pragma unroll
      for (int i = 0; i < 4; ++i) {
        const int m = wm * 64 + i * 16 + l16;
        af[i] = *(const v16bf*)(As + m * 32 + klo * 2);
      }
    }
    v16bf bfr[2];
#pragma unroll
    for (int j = 0; j < 2; ++j) {
      const int n = wn * 32 + j * 16 + l16;
      bfr[j] = *(const v16bf*)(Bst + n * 32 + half * 16);
    }
#pragma unroll
    for (int i = 0; i < 4; ++i)
#pragma unroll
      for (int j = 0; j < 2; ++j)
        acc[i][j] = __builtin_amdgcn_wmma_f32_16x16x32_bf16(
            false, af[i], false, bfr[j], (short)0, acc[i][j], false, false);
    __syncthreads();
  }

  // ---- epilogue: bias add + store ----
#pragma unroll
  for (int i = 0; i < 4; ++i) {
#pragma unroll
    for (int j = 0; j < 2; ++j) {
      const int n = bn0 + wn * 32 + j * 16 + l16;
      const float bv = bias[n];
#pragma unroll
      for (int r = 0; r < 8; ++r) {
        const int m = bm0 + wm * 64 + i * 16 + r + half * 8;
        store_out(C + (size_t)m * N + n, acc[i][j][r] + bv);
      }
    }
  }
}

// ---------------------------------------------------------------------------
// Pack video qkv: window gather + RMSNorm(q,k) + bf16, layout [win][head][l][d]
// grid = NWIN*WLEN blocks, 256 threads: 16 threads per head, 8 dims each.
// ---------------------------------------------------------------------------
__global__ __launch_bounds__(256)
void pack_video_kernel(const bf16_t* __restrict__ qkv,
                       const float* __restrict__ gq, const float* __restrict__ gk,
                       bf16_t* __restrict__ Qv, bf16_t* __restrict__ Kv,
                       bf16_t* __restrict__ Vv) {
  const int b   = blockIdx.x;
  const int win = b >> 8;
  const int l   = b & 255;
  const int nt = win >> 4, nh = (win >> 2) & 3, nw = win & 3;
  const int tt = l >> 6,  hh = (l >> 3) & 7,  ww = l & 7;
  const size_t srow = ((size_t)(nt * WT_ + tt) * H_ + (nh * WH_ + hh)) * W_ + (nw * WW_ + ww);

  const int tid  = threadIdx.x;
  const int head = tid >> 4;
  const int s16  = tid & 15;
  const bf16_t* base = qkv + srow * DIM3;
  const size_t orow = ((size_t)(win * HEADS + head) * WLEN + l) * HD;

  // q (rmsnorm with gq)
  {
    const bf16_t* src = base + head * HD + s16 * 8;
    float v[8]; float ss = 0.f;
#pragma unroll
    for (int i = 0; i < 8; ++i) { v[i] = bf2f(src[i]); ss += v[i] * v[i]; }
    ss += __shfl_xor(ss, 1, 32); ss += __shfl_xor(ss, 2, 32);
    ss += __shfl_xor(ss, 4, 32); ss += __shfl_xor(ss, 8, 32);
    const float rr = rsqrtf(ss * (1.f / HD) + 1e-6f);
#pragma unroll
    for (int i = 0; i < 8; ++i)
      Qv[orow + s16 * 8 + i] = f2bf(v[i] * rr * gq[s16 * 8 + i]);
  }
  // k (rmsnorm with gk)
  {
    const bf16_t* src = base + DIM + head * HD + s16 * 8;
    float v[8]; float ss = 0.f;
#pragma unroll
    for (int i = 0; i < 8; ++i) { v[i] = bf2f(src[i]); ss += v[i] * v[i]; }
    ss += __shfl_xor(ss, 1, 32); ss += __shfl_xor(ss, 2, 32);
    ss += __shfl_xor(ss, 4, 32); ss += __shfl_xor(ss, 8, 32);
    const float rr = rsqrtf(ss * (1.f / HD) + 1e-6f);
#pragma unroll
    for (int i = 0; i < 8; ++i)
      Kv[orow + s16 * 8 + i] = f2bf(v[i] * rr * gk[s16 * 8 + i]);
  }
  // v (plain copy)
  {
    const bf16_t* src = base + 2 * DIM + head * HD + s16 * 8;
#pragma unroll
    for (int i = 0; i < 8; ++i) Vv[orow + s16 * 8 + i] = src[i];
  }
}

// ---------------------------------------------------------------------------
// Pack text qkv: RMSNorm(q,k), layout [head][row][d].  grid = LT blocks.
// ---------------------------------------------------------------------------
__global__ __launch_bounds__(256)
void pack_text_kernel(const bf16_t* __restrict__ qkv,
                      const float* __restrict__ gq, const float* __restrict__ gk,
                      bf16_t* __restrict__ Qt, bf16_t* __restrict__ Kt,
                      bf16_t* __restrict__ Vt) {
  const int row  = blockIdx.x;
  const int tid  = threadIdx.x;
  const int head = tid >> 4;
  const int s16  = tid & 15;
  const bf16_t* base = qkv + (size_t)row * DIM3;
  const size_t orow = ((size_t)head * LT + row) * HD;

  {
    const bf16_t* src = base + head * HD + s16 * 8;
    float v[8]; float ss = 0.f;
#pragma unroll
    for (int i = 0; i < 8; ++i) { v[i] = bf2f(src[i]); ss += v[i] * v[i]; }
    ss += __shfl_xor(ss, 1, 32); ss += __shfl_xor(ss, 2, 32);
    ss += __shfl_xor(ss, 4, 32); ss += __shfl_xor(ss, 8, 32);
    const float rr = rsqrtf(ss * (1.f / HD) + 1e-6f);
#pragma unroll
    for (int i = 0; i < 8; ++i)
      Qt[orow + s16 * 8 + i] = f2bf(v[i] * rr * gq[s16 * 8 + i]);
  }
  {
    const bf16_t* src = base + DIM + head * HD + s16 * 8;
    float v[8]; float ss = 0.f;
#pragma unroll
    for (int i = 0; i < 8; ++i) { v[i] = bf2f(src[i]); ss += v[i] * v[i]; }
    ss += __shfl_xor(ss, 1, 32); ss += __shfl_xor(ss, 2, 32);
    ss += __shfl_xor(ss, 4, 32); ss += __shfl_xor(ss, 8, 32);
    const float rr = rsqrtf(ss * (1.f / HD) + 1e-6f);
#pragma unroll
    for (int i = 0; i < 8; ++i)
      Kt[orow + s16 * 8 + i] = f2bf(v[i] * rr * gk[s16 * 8 + i]);
  }
  {
    const bf16_t* src = base + 2 * DIM + head * HD + s16 * 8;
#pragma unroll
    for (int i = 0; i < 8; ++i) Vt[orow + s16 * 8 + i] = src[i];
  }
}

// ---------------------------------------------------------------------------
// Flash attention.  One block per (win, head, 128-query slab).  8 waves, each
// owns 16 query rows.  Keys streamed 64 at a time: K tile via the Tensor Data
// Mover (TENSOR_LOAD_TO_LDS + s_wait_tensorcnt), V transposed manually.
// WMMA bf16 for QK^T and PV; online softmax; P bounced through per-wave LDS.
// Dynamic LDS layout: [0,16K) K tile, [16K,32K) V^T tile, [32K,48K) P tiles.
// ---------------------------------------------------------------------------
__global__ __launch_bounds__(256)
void attn_kernel(const bf16_t* __restrict__ Qv, const bf16_t* __restrict__ Kv,
                 const bf16_t* __restrict__ Vv, const bf16_t* __restrict__ Qt,
                 const bf16_t* __restrict__ Kt, const bf16_t* __restrict__ Vt,
                 bf16_t* __restrict__ vo_in, float* __restrict__ to_acc) {
  extern __shared__ char smem_raw[];
  bf16_t* Ks  = (bf16_t*)smem_raw;                  // [key][dim]   64*128
  bf16_t* Vts = (bf16_t*)(smem_raw + 16 * 1024);    // [dim][key]   128*64
  bf16_t* Psb = (bf16_t*)(smem_raw + 32 * 1024);    // 8 waves x [16][64]

  const int qb   = blockIdx.x & 3;
  const int head = (blockIdx.x >> 2) & 15;
  const int win  = blockIdx.x >> 6;
  const int tid  = threadIdx.x;
  const int wave = tid >> 5;
  const int lane = tid & 31;
  const int l16  = lane & 15;
  const int half = lane >> 4;
  const int q0   = qb * 128 + wave * 16;        // 0..511
  const float scale = 0.088388347648318447f;    // 1/sqrt(128)

  // ---- load Q fragments (held in registers for whole kernel) ----
  const bf16_t* Qbase = (q0 < WLEN)
      ? (Qv + ((size_t)(win * HEADS + head) * WLEN + q0) * HD)
      : (Qt + ((size_t)head * LT + (q0 - WLEN)) * HD);
  v16bf qf[4];
  {
    const bf16_t* qrow = Qbase + (size_t)l16 * HD;
    const int klo = half * 8;
#pragma unroll
    for (int c = 0; c < 4; ++c) {
      qf[c] = *(const v16bf*)(qrow + c * 32 + klo * 2);
    }
  }

  v8f of[8];
#pragma unroll
  for (int dg = 0; dg < 8; ++dg)
#pragma unroll
    for (int r = 0; r < 8; ++r) of[dg][r] = 0.0f;
  float mrow[8], lrow[8];
#pragma unroll
  for (int r = 0; r < 8; ++r) { mrow[r] = -1e30f; lrow[r] = 0.0f; }

  for (int kb = 0; kb < 8; ++kb) {
    const int key0 = kb * 64;
    const bf16_t* Kbase = (key0 < WLEN)
        ? (Kv + ((size_t)(win * HEADS + head) * WLEN + key0) * HD)
        : (Kt + ((size_t)head * LT + (key0 - WLEN)) * HD);
    const bf16_t* Vbase = (key0 < WLEN)
        ? (Vv + ((size_t)(win * HEADS + head) * WLEN + key0) * HD)
        : (Vt + ((size_t)head * LT + (key0 - WLEN)) * HD);

    // ---- K tile via Tensor Data Mover (async DMA global -> LDS) ----
    if (wave == 0) {
      tdm_load_2d_to_lds(/*lds off*/ 0u, Kbase, /*row elems*/ HD,
                         /*rows*/ 64, /*stride*/ HD);
      __builtin_amdgcn_s_wait_tensorcnt(0);
    }
    // ---- V tile transposed into LDS (lane-level transpose) ----
    {
      const int row = tid >> 2;            // 0..63
      const int dq  = (tid & 3) * 32;      // dim quarter
      const bf16_t* vsrc = Vbase + (size_t)row * HD + dq;
#pragma unroll
      for (int d = 0; d < 32; ++d)
        Vts[(dq + d) * 64 + row] = vsrc[d];
    }
    __syncthreads();

    // ---- S = scale * Q K^T for 4 sub-tiles of 16 keys ----
    v8f sf[4];
#pragma unroll
    for (int t = 0; t < 4; ++t) {
#pragma unroll
      for (int r = 0; r < 8; ++r) sf[t][r] = 0.0f;
#pragma unroll
      for (int c = 0; c < 4; ++c) {
        v16bf bfr = *(const v16bf*)(Ks + (t * 16 + l16) * HD + c * 32 + half * 16);
        sf[t] = __builtin_amdgcn_wmma_f32_16x16x32_bf16(
            false, qf[c], false, bfr, (short)0, sf[t], false, false);
      }
#pragma unroll
      for (int r = 0; r < 8; ++r) sf[t][r] = sf[t][r] * scale;
    }

    // ---- online softmax: row max over 64 keys ----
    float mnew[8];
#pragma unroll
    for (int r = 0; r < 8; ++r) {
      float v = fmaxf(fmaxf(sf[0][r], sf[1][r]), fmaxf(sf[2][r], sf[3][r]));
      v = fmaxf(v, __shfl_xor(v, 1, 32));
      v = fmaxf(v, __shfl_xor(v, 2, 32));
      v = fmaxf(v, __shfl_xor(v, 4, 32));
      v = fmaxf(v, __shfl_xor(v, 8, 32));
      mnew[r] = fmaxf(mrow[r], v);
    }
#pragma unroll
    for (int r = 0; r < 8; ++r) {
      const float alpha = __expf(mrow[r] - mnew[r]);
      mrow[r] = mnew[r];
      lrow[r] *= alpha;
#pragma unroll
      for (int dg = 0; dg < 8; ++dg) of[dg][r] = of[dg][r] * alpha;
    }

    // ---- P = exp(S - m), write bf16 to per-wave LDS tile, row sums ----
    bf16_t* pw = Psb + wave * (16 * 64);
#pragma unroll
    for (int r = 0; r < 8; ++r) {
      float rs = 0.f;
#pragma unroll
      for (int t = 0; t < 4; ++t) {
        const float p = __expf(sf[t][r] - mrow[r]);
        rs += p;
        pw[(r + half * 8) * 64 + t * 16 + l16] = f2bf(p);
      }
      rs += __shfl_xor(rs, 1, 32); rs += __shfl_xor(rs, 2, 32);
      rs += __shfl_xor(rs, 4, 32); rs += __shfl_xor(rs, 8, 32);
      lrow[r] += rs;
    }

    // ---- reload P as A-fragments, O += P V ----
    v16bf pf[2];
    {
      const int klo = half * 8;
#pragma unroll
      for (int pc = 0; pc < 2; ++pc) {
        pf[pc] = *(const v16bf*)(pw + l16 * 64 + pc * 32 + klo * 2);
      }
    }
#pragma unroll
    for (int dg = 0; dg < 8; ++dg) {
#pragma unroll
      for (int pc = 0; pc < 2; ++pc) {
        v16bf vb = *(const v16bf*)(Vts + (dg * 16 + l16) * 64 + pc * 32 + half * 16);
        of[dg] = __builtin_amdgcn_wmma_f32_16x16x32_bf16(
            false, pf[pc], false, vb, (short)0, of[dg], false, false);
      }
    }
    __syncthreads();
  }

  // ---- finalize: O /= l, write out ----
  float inv[8];
#pragma unroll
  for (int r = 0; r < 8; ++r) inv[r] = 1.0f / lrow[r];

  if (q0 < WLEN) {
    const int nt = win >> 4, nh = (win >> 2) & 3, nw = win & 3;
#pragma unroll
    for (int r = 0; r < 8; ++r) {
      const int lidx = q0 + r + half * 8;
      const int tt = lidx >> 6, hh = (lidx >> 3) & 7, ww = lidx & 7;
      const size_t srow =
          ((size_t)(nt * WT_ + tt) * H_ + (nh * WH_ + hh)) * W_ + (nw * WW_ + ww);
      bf16_t* dst = vo_in + srow * DIM + head * HD;
#pragma unroll
      for (int dg = 0; dg < 8; ++dg)
        dst[dg * 16 + l16] = f2bf(of[dg][r] * inv[r]);
    }
  } else {
    const float wmean = 1.0f / (float)NWIN;
#pragma unroll
    for (int r = 0; r < 8; ++r) {
      const int trow = q0 - WLEN + r + half * 8;
      float* dst = to_acc + (size_t)trow * DIM + head * HD;
#pragma unroll
      for (int dg = 0; dg < 8; ++dg)
        atomicAdd(&dst[dg * 16 + l16], of[dg][r] * inv[r] * wmean);
    }
  }
}

// ---------------------------------------------------------------------------
// to_in (bf16) = to_acc (fp32 mean-accumulated)
// ---------------------------------------------------------------------------
__global__ __launch_bounds__(256)
void finalize_text_kernel(const float* __restrict__ acc, bf16_t* __restrict__ out,
                          int n) {
  const int i = blockIdx.x * 256 + threadIdx.x;
  if (i < n) out[i] = f2bf(acc[i]);
}

// ---------------------------------------------------------------------------
// Host orchestration
// ---------------------------------------------------------------------------
extern "C" void kernel_launch(void* const* d_in, const int* in_sizes, int n_in,
                              void* d_out, int out_size, void* d_ws, size_t ws_size,
                              hipStream_t stream) {
  (void)in_sizes; (void)n_in; (void)out_size; (void)ws_size;

  const float* vid    = (const float*)d_in[0];
  const float* txt    = (const float*)d_in[1];
  const float* Wqkv_v = (const float*)d_in[2];
  const float* bqkv_v = (const float*)d_in[3];
  const float* Wqkv_t = (const float*)d_in[4];
  const float* bqkv_t = (const float*)d_in[5];
  const float* Wo_v   = (const float*)d_in[6];
  const float* bo_v   = (const float*)d_in[7];
  const float* Wo_t   = (const float*)d_in[8];
  const float* bo_t   = (const float*)d_in[9];
  const float* gq_v   = (const float*)d_in[10];
  const float* gk_v   = (const float*)d_in[11];
  const float* gq_t   = (const float*)d_in[12];
  const float* gk_t   = (const float*)d_in[13];

  char* ws = (char*)d_ws;
  size_t off = 0;
  auto alloc = [&](size_t bytes) -> char* {
    char* p = ws + off;
    off += (bytes + 255) & ~(size_t)255;
    return p;
  };

  bf16_t* qkv_v  = (bf16_t*)alloc((size_t)LV * DIM3 * 2);
  bf16_t* qkv_t  = (bf16_t*)alloc((size_t)LT * DIM3 * 2);
  bf16_t* Qv     = (bf16_t*)alloc((size_t)NWIN * HEADS * WLEN * HD * 2);
  bf16_t* Kv     = (bf16_t*)alloc((size_t)NWIN * HEADS * WLEN * HD * 2);
  bf16_t* Vv     = (bf16_t*)alloc((size_t)NWIN * HEADS * WLEN * HD * 2);
  bf16_t* Qt     = (bf16_t*)alloc((size_t)HEADS * LT * HD * 2);
  bf16_t* Kt     = (bf16_t*)alloc((size_t)HEADS * LT * HD * 2);
  bf16_t* Vt     = (bf16_t*)alloc((size_t)HEADS * LT * HD * 2);
  bf16_t* vo_in  = (bf16_t*)alloc((size_t)LV * DIM * 2);
  bf16_t* to_in  = (bf16_t*)alloc((size_t)LT * DIM * 2);
  float*  to_acc = (float*)alloc((size_t)LT * DIM * 4);

  float* vo_out = (float*)d_out;
  float* to_out = (float*)d_out + (size_t)LV * DIM;

  // 1) QKV GEMMs (fp32 inputs -> bf16 accum-out)
  gemm_bias_kernel<float, bf16_t><<<dim3(DIM3 / 128, LV / 128), 256, 0, stream>>>(
      vid, Wqkv_v, bqkv_v, qkv_v, LV, DIM3, DIM);
  gemm_bias_kernel<float, bf16_t><<<dim3(DIM3 / 128, LT / 128), 256, 0, stream>>>(
      txt, Wqkv_t, bqkv_t, qkv_t, LT, DIM3, DIM);

  // 2) RMSNorm + window pack
  pack_video_kernel<<<NWIN * WLEN, 256, 0, stream>>>(qkv_v, gq_v, gk_v, Qv, Kv, Vv);
  pack_text_kernel<<<LT, 256, 0, stream>>>(qkv_t, gq_t, gk_t, Qt, Kt, Vt);

  // 3) zero text accumulator, run flash attention (48KB dynamic LDS)
  hipMemsetAsync(to_acc, 0, (size_t)LT * DIM * 4, stream);
  attn_kernel<<<NWIN * HEADS * 4, 256, 48 * 1024, stream>>>(Qv, Kv, Vv, Qt, Kt, Vt,
                                                            vo_in, to_acc);

  // 4) text mean -> bf16
  finalize_text_kernel<<<(LT * DIM + 255) / 256, 256, 0, stream>>>(
      to_acc, to_in, LT * DIM);

  // 5) output projections (bf16 A, fp32 out into d_out)
  gemm_bias_kernel<bf16_t, float><<<dim3(DIM / 128, LV / 128), 256, 0, stream>>>(
      vo_in, Wo_v, bo_v, vo_out, LV, DIM, DIM);
  gemm_bias_kernel<bf16_t, float><<<dim3(DIM / 128, LT / 128), 256, 0, stream>>>(
      to_in, Wo_t, bo_t, to_out, LT, DIM, DIM);
}
